// CoverageLoss_72756745994379
// MI455X (gfx1250) — compile-verified, matrix operand order checked by reading, not run on hardware
//
#include <hip/hip_runtime.h>
#include <hip/hip_bf16.h>
#include <float.h>

// ---------------------------------------------------------------------------
// CoverageLoss, algebraically reduced:
//   TAIL_K == PUSH_K == 4 and empty_dists rows == space_dists[far_inds] rows,
//   so close_dists == tail_dists[far_inds]. Output = mean(huber(top64(tail))).
// tail[m] = mean of 4 smallest L1 dists between space_samples[m] and latents.
//
// Distance core uses V_WMMA_F32_16X16X32_F16 with B = ones to reduce 16 rows
// x 32 k of packed-f16 |a-b| per instruction, accumulating in f32.
// ---------------------------------------------------------------------------

typedef __attribute__((ext_vector_type(16))) _Float16     v16h;
typedef __attribute__((ext_vector_type(8)))  float        v8f;
typedef __attribute__((ext_vector_type(8)))  unsigned int v8u;

#define M_ROWS   2048
#define N_ROWS   8192
#define DIMS     256
#define ROW_DW   128          // 256 f16 packed as 128 dwords
#define NSUB     32           // n-subslices
#define N_PER    (N_ROWS / NSUB)   // 256 n per wave

// workspace layout (bytes)
#define OFF_SP16   ((size_t)0)                       // 2048*256*2 = 1 MB
#define OFF_LAT16  ((size_t)(1u << 20))              // 8192*256*2 = 4 MB
#define OFF_PART   (OFF_LAT16 + ((size_t)4u << 20))  // 2048*32*4 f32 = 1 MB
#define OFF_TAIL   (OFF_PART + ((size_t)1u << 20))   // 2048 f32

__device__ __forceinline__ void ins4(float d, float4& m) {
    // branchless insert into ascending 4-smallest list
    float t = d, a;
    a = fminf(t, m.x); t = fmaxf(t, m.x); m.x = a;
    a = fminf(t, m.y); t = fmaxf(t, m.y); m.y = a;
    a = fminf(t, m.z); t = fmaxf(t, m.z); m.z = a;
    m.w = fminf(t, m.w);
}

__device__ __forceinline__ float huberf(float x) {
    float ax = fabsf(x);
    return (ax < 1.0f) ? 0.5f * x * x : ax - 0.5f;
}

// ---- kernel 0: f32 -> packed f16 dwords ------------------------------------
__global__ __launch_bounds__(256) void conv_f16_kernel(
        const float* __restrict__ src, unsigned int* __restrict__ dst, int ndw) {
    int i = blockIdx.x * blockDim.x + threadIdx.x;
    if (i < ndw) {
        _Float16 h0 = (_Float16)src[2 * i];
        _Float16 h1 = (_Float16)src[2 * i + 1];
        unsigned int lo = (unsigned int)__builtin_bit_cast(unsigned short, h0);
        unsigned int hi = (unsigned int)__builtin_bit_cast(unsigned short, h1);
        dst[i] = lo | (hi << 16);
    }
}

// ---- kernel 1: WMMA L1-distance + per-wave 4-smallest ----------------------
// grid (128, 4), block 256 (8 waves). Each wave: 16 a-rows x 256 n.
__global__ __launch_bounds__(256) void dist_tail_kernel(
        const unsigned int* __restrict__ sp16,
        const unsigned int* __restrict__ lat16,
        float* __restrict__ part) {
    const int lane = threadIdx.x & 31;
    const int wave = threadIdx.x >> 5;
    const int m    = lane & 15;     // A-fragment row owned by this lane
    const int half = lane >> 4;     // lane half selects k +8 per ISA layout
    const int tile = blockIdx.x;                 // 0..127 (16 rows each)
    const int sub  = blockIdx.y * 8 + wave;      // 0..31
    const int rowBase = tile * 16;
    const int n0 = sub * N_PER;

    // Preload A fragments for all 8 k-chunks: lane needs dwords
    //   rowDW[c*16 + half*4 + {0..3}] and rowDW[c*16 + 8 + half*4 + {0..3}]
    // matching the 16-bit A 16x32 VGPR layout (K = 2(j&3) + 16*(j>=4) + 8*half).
    const unsigned int* arow = sp16 + (size_t)(rowBase + m) * ROW_DW;
    v8u af[8];
#pragma unroll
    for (int c = 0; c < 8; ++c) {
        uint4 lo = *(const uint4*)(arow + c * 16 + half * 4);
        uint4 hi = *(const uint4*)(arow + c * 16 + 8 + half * 4);
        v8u u = {lo.x, lo.y, lo.z, lo.w, hi.x, hi.y, hi.z, hi.w};
        af[c] = u;
    }

    v16h ones;
#pragma unroll
    for (int i = 0; i < 16; ++i) ones[i] = (_Float16)1.0f;

    float4 mn[8];
#pragma unroll
    for (int r = 0; r < 8; ++r) mn[r] = make_float4(3.0e38f, 3.0e38f, 3.0e38f, 3.0e38f);

    for (int n = n0; n < n0 + N_PER; ++n) {
        const unsigned int* brow = lat16 + (size_t)n * ROW_DW;
        v8f acc0 = {};
        v8f acc1 = {};
#pragma unroll
        for (int c = 0; c < 8; ++c) {
            uint4 lo = *(const uint4*)(brow + c * 16 + half * 4);
            uint4 hi = *(const uint4*)(brow + c * 16 + 8 + half * 4);
            v8u bu = {lo.x, lo.y, lo.z, lo.w, hi.x, hi.y, hi.z, hi.w};
            v16h av = __builtin_bit_cast(v16h, af[c]);
            v16h bv = __builtin_bit_cast(v16h, bu);
            v16h dv = av - bv;                       // packed f16 subtract
            v8u  du = __builtin_bit_cast(v8u, dv);
            du &= 0x7FFF7FFFu;                       // |.| = clear sign bits
            v16h ad = __builtin_bit_cast(v16h, du);
            // D[m,:] += sum_k |a[m,k]-b[n,k]|  (B = ones, f32 accumulate)
            if (c & 1)
                acc1 = __builtin_amdgcn_wmma_f32_16x16x32_f16(
                        false, ad, false, ones, (short)0, acc1, false, false);
            else
                acc0 = __builtin_amdgcn_wmma_f32_16x16x32_f16(
                        false, ad, false, ones, (short)0, acc0, false, false);
        }
        v8f d = acc0 + acc1;   // lane holds rows (half*8 + r), replicated over columns
#pragma unroll
        for (int r = 0; r < 8; ++r) ins4(d[r], mn[r]);
    }

    // lanes 0 (rows 0..7) and 16 (rows 8..15) publish partial min-4 lists
    if (m == 0) {
        float4* p4 = (float4*)part;
#pragma unroll
        for (int r = 0; r < 8; ++r) {
            int row = rowBase + half * 8 + r;
            p4[(size_t)row * NSUB + sub] = mn[r];
        }
    }
}

// ---- kernel 2: merge partial min-4 lists -> tail[row] ----------------------
__global__ __launch_bounds__(256) void tail_reduce_kernel(
        const float4* __restrict__ part, float* __restrict__ tail) {
    int r = blockIdx.x * blockDim.x + threadIdx.x;
    if (r >= M_ROWS) return;
    float4 m = make_float4(3.0e38f, 3.0e38f, 3.0e38f, 3.0e38f);
    for (int s = 0; s < NSUB; ++s) {
        float4 v = part[(size_t)r * NSUB + s];
        ins4(v.x, m); ins4(v.y, m); ins4(v.z, m); ins4(v.w, m);
    }
    tail[r] = (m.x + m.y + m.z + m.w) * 0.25f;
}

// ---- kernel 3: top-64 extraction + huber mean ------------------------------
__global__ __launch_bounds__(256) void topk_huber_kernel(
        const float* __restrict__ tail, float* __restrict__ out) {
    __shared__ float sv[M_ROWS];
    __shared__ float red[256];
    __shared__ int   redi[256];
    for (int j = 0; j < 8; ++j)
        sv[threadIdx.x + 256 * j] = tail[threadIdx.x + 256 * j];
    __syncthreads();
    float sum = 0.0f;
    for (int t = 0; t < 64; ++t) {
        float bm = -3.0e38f; int bi = 0;
        for (int j = 0; j < 8; ++j) {
            int i = threadIdx.x + 256 * j;
            float v = sv[i];
            if (v > bm) { bm = v; bi = i; }
        }
        red[threadIdx.x] = bm; redi[threadIdx.x] = bi;
        __syncthreads();
        if (threadIdx.x == 0) {
            float gm = -3.0e38f; int gi = 0;
            for (int j = 0; j < 256; ++j)
                if (red[j] > gm) { gm = red[j]; gi = redi[j]; }
            sum += huberf(gm);
            sv[gi] = -3.0e38f;
        }
        __syncthreads();
    }
    if (threadIdx.x == 0) out[0] = sum * (1.0f / 64.0f);
}

// ---------------------------------------------------------------------------
extern "C" void kernel_launch(void* const* d_in, const int* in_sizes, int n_in,
                              void* d_out, int out_size, void* d_ws, size_t ws_size,
                              hipStream_t stream) {
    const float* space = (const float*)d_in[0];   // [2048, 256] f32
    const float* lat   = (const float*)d_in[1];   // [8192, 256] f32
    char* ws = (char*)d_ws;
    unsigned int* sp16  = (unsigned int*)(ws + OFF_SP16);
    unsigned int* lat16 = (unsigned int*)(ws + OFF_LAT16);
    float* part = (float*)(ws + OFF_PART);
    float* tail = (float*)(ws + OFF_TAIL);
    float* out  = (float*)d_out;

    const int sp_dw  = M_ROWS * ROW_DW;   // 262144
    const int lat_dw = N_ROWS * ROW_DW;   // 1048576

    conv_f16_kernel<<<(sp_dw + 255) / 256, 256, 0, stream>>>(space, sp16, sp_dw);
    conv_f16_kernel<<<(lat_dw + 255) / 256, 256, 0, stream>>>(lat, lat16, lat_dw);
    dist_tail_kernel<<<dim3(M_ROWS / 16, NSUB / 8, 1), 256, 0, stream>>>(sp16, lat16, part);
    tail_reduce_kernel<<<M_ROWS / 256, 256, 0, stream>>>((const float4*)part, tail);
    topk_huber_kernel<<<1, 256, 0, stream>>>(tail, out);
}
